// ParallelDroplessMLP_76708115906967
// MI455X (gfx1250) — matrix operand order repaired and checked by reading, not compiled, not run on hardware
//
#include <hip/hip_runtime.h>
#include <hip/hip_bf16.h>
#include <math.h>

#define SLEN 2048
#define BSZ  4
#define HSZ  1024
#define FFN  4096
#define NEXP 8
#define TOPK 2
#define NTOK (SLEN * BSZ)      /* 8192 tokens */
#define CAP  (NTOK * TOPK)     /* worst-case rows per expert */
#define MT   32                /* rows (tokens) per block: two 16-row WMMA tiles */
#define XSTR 1032              /* padded LDS row stride (bf16 elems) for X   */
#define HSTR 136               /* padded LDS row stride (bf16 elems) for h   */
#define FCHK 128               /* FFN chunk width per iteration              */
#define TILE 512               /* elems per 32x16 bf16 B tile (1KB)          */

typedef __attribute__((ext_vector_type(16))) __bf16 bf16x16;
typedef __attribute__((ext_vector_type(8)))  __bf16 bf16x8;
typedef __attribute__((ext_vector_type(8)))  float  f32x8;

static __device__ __forceinline__ unsigned short f2bf(float x) {
    unsigned u = __float_as_uint(x);
    u += 0x7FFFu + ((u >> 16) & 1u);         // round-to-nearest-even
    return (unsigned short)(u >> 16);
}

static __device__ __forceinline__ unsigned pack2(float a, float b) {
    return (unsigned)f2bf(a) | ((unsigned)f2bf(b) << 16);
}

static __device__ __forceinline__ bf16x16 cat8(bf16x8 lo, bf16x8 hi) {
    return __builtin_shufflevector(lo, hi, 0,1,2,3,4,5,6,7,8,9,10,11,12,13,14,15);
}

// A operand (16x32 bf16) from LDS row-major [16][stride], K-offset kk.
// ISA 7.12.2: lanes 0-15 -> M=lane, K=kk+0..7 & kk+16..23; lanes 16-31 -> K=kk+8..15 & kk+24..31.
static __device__ __forceinline__ bf16x16 load_a(const unsigned short* lds, int stride,
                                                 int kk, int lane) {
    int m  = lane & 15;
    int kb = (lane & 16) ? 8 : 0;
    const unsigned short* p = lds + m * stride + kk + kb;
    bf16x8 lo = *(const bf16x8*)(p);        // ds_load_b128
    bf16x8 hi = *(const bf16x8*)(p + 16);
    return cat8(lo, hi);
}

// B operand (32x16 bf16) from a pre-swizzled 1KB tile: lane-dense, fully coalesced.
// Tile layout: [granule g(2)][lane L(32)][8 bf16]; lane L holds row K=L, cols g*8..g*8+7.
static __device__ __forceinline__ bf16x16 load_bt(const unsigned short* tb, int lane) {
    int n  = lane & 15;
    int kb = (lane & 16) ? 8 : 0;
    const unsigned short* p = tb + (n >> 3) * 256 + (n & 7);
    bf16x16 v;
    #pragma unroll
    for (int s = 0; s < 16; ++s) {
        int k = kb + (s & 7) + ((s & 8) ? 16 : 0);
        v[s] = __builtin_bit_cast(__bf16, p[k * 8]);
    }
    return v;
}

// ------------------------------------------------- weight cvt + tile swizzle
// w1 f32 [E][HS][FFN] -> bf16 tiles [(e*256+nb)*32+kb] (K fastest; nb=n/16, kb=k/32)
__global__ void cvt_swz_w1_kernel(const float* __restrict__ src,
                                  unsigned short* __restrict__ dst) {
    const int total = NEXP * HSZ * (FFN / 8);
    int i = blockIdx.x * blockDim.x + threadIdx.x;
    int stride = gridDim.x * blockDim.x;
    for (; i < total; i += stride) {
        int e   = i / (HSZ * (FFN / 8));
        int rem = i % (HSZ * (FFN / 8));
        int k   = rem / (FFN / 8);
        int n8  = rem % (FFN / 8);
        const float* s = src + ((size_t)e * HSZ + k) * FFN + n8 * 8;
        float4 f0 = *(const float4*)(s);
        float4 f1 = *(const float4*)(s + 4);
        int nb = n8 >> 1, g = n8 & 1, kb = k >> 5, L = k & 31;
        size_t o = ((((size_t)e * 256 + nb) * 32 + kb) * TILE) + g * 256 + L * 8;
        uint4 u;
        u.x = pack2(f0.x, f0.y); u.y = pack2(f0.z, f0.w);
        u.z = pack2(f1.x, f1.y); u.w = pack2(f1.z, f1.w);
        *(uint4*)(dst + o) = u;
    }
}

// w2 f32 [E][FFN][HS] -> bf16 tiles [(e*128+kb)*64+nb] (N fastest; kb=k/32, nb=n/16)
__global__ void cvt_swz_w2_kernel(const float* __restrict__ src,
                                  unsigned short* __restrict__ dst) {
    const int total = NEXP * FFN * (HSZ / 8);
    int i = blockIdx.x * blockDim.x + threadIdx.x;
    int stride = gridDim.x * blockDim.x;
    for (; i < total; i += stride) {
        int e   = i / (FFN * (HSZ / 8));
        int rem = i % (FFN * (HSZ / 8));
        int k   = rem / (HSZ / 8);
        int n8  = rem % (HSZ / 8);
        const float* s = src + ((size_t)e * FFN + k) * HSZ + n8 * 8;
        float4 f0 = *(const float4*)(s);
        float4 f1 = *(const float4*)(s + 4);
        int nb = n8 >> 1, g = n8 & 1, kb = k >> 5, L = k & 31;
        size_t o = ((((size_t)e * 128 + kb) * 64 + nb) * TILE) + g * 256 + L * 8;
        uint4 u;
        u.x = pack2(f0.x, f0.y); u.y = pack2(f0.z, f0.w);
        u.z = pack2(f1.x, f1.y); u.w = pack2(f1.z, f1.w);
        *(uint4*)(dst + o) = u;
    }
}

// ---------------------------------------------------------------- routing
__global__ void route_kernel(const int* __restrict__ eidx, const float* __restrict__ ew,
                             int* __restrict__ counts, int* __restrict__ toks,
                             float* __restrict__ cws) {
    int i = blockIdx.x * blockDim.x + threadIdx.x;
    if (i >= NTOK * TOPK) return;
    int e = eidx[i];
    if (e < 0) e = 0;
    if (e >= NEXP) e = NEXP - 1;
    int pos = atomicAdd(counts + e, 1);
    toks[e * CAP + pos] = i / TOPK;
    cws [e * CAP + pos] = ew[i];
}

// ---------------------------------------------------------------- fused MoE MLP
// block = (expert e, tile of MT=32 routed rows). 8 waves: wave w owns h cols
// [w*16,w*16+16) in GEMM1 and out cols [w*128,(w+1)*128) in GEMM2.
// B tiles are pre-swizzled 1KB blocks -> every load is lane-dense & sequential.
__global__ __launch_bounds__(256) void moe_mlp_kernel(
    const float* __restrict__ x,
    const int* __restrict__ counts,
    const int* __restrict__ toks,
    const float* __restrict__ cws,
    const unsigned short* __restrict__ w1t,   // swizzled tiles
    const unsigned short* __restrict__ w2t,   // swizzled tiles
    float* __restrict__ out)                  // [NTOK][HS] f32 (pre-zeroed)
{
    __shared__ unsigned short Xlds[MT * XSTR];   // gathered tokens, bf16 (~66KB)
    __shared__ unsigned short Hlds[MT * HSTR];   // gelu(X@w1) chunk, bf16 (~8.7KB)
    __shared__ int   tokS[MT];
    __shared__ float cwS[MT];

    const int e    = blockIdx.x;
    const int base = blockIdx.y * MT;
    const int cnt  = counts[e];
    if (base >= cnt) return;

    const int tid  = threadIdx.x;
    const int lane = tid & 31;
    const int wv   = tid >> 5;

    if (tid < MT) {
        int r = base + tid;
        if (r < cnt) { tokS[tid] = toks[e * CAP + r]; cwS[tid] = cws[e * CAP + r]; }
        else         { tokS[tid] = 0;                 cwS[tid] = 0.0f; }
    }
    __syncthreads();

    // gather X rows: thread handles 128 elems of one row (f32 -> bf16 -> LDS)
    {
        int r   = tid >> 3;             // 0..31
        int seg = (tid & 7) * 128;      // 0..896
        const float* src = x + (size_t)tokS[r] * HSZ + seg;
        unsigned short* dst = Xlds + r * XSTR + seg;
        #pragma unroll
        for (int j = 0; j < 128; j += 4) {
            float4 f = *(const float4*)(src + j);
            ushort4 b;
            b.x = f2bf(f.x); b.y = f2bf(f.y); b.z = f2bf(f.z); b.w = f2bf(f.w);
            *(ushort4*)(dst + j) = b;
        }
    }
    __syncthreads();

    const f32x8 zero = {0.f,0.f,0.f,0.f,0.f,0.f,0.f,0.f};
    f32x8 accOut[2][8];
    #pragma unroll
    for (int h2 = 0; h2 < 2; ++h2)
        #pragma unroll
        for (int i = 0; i < 8; ++i) accOut[h2][i] = zero;

    for (int fc = 0; fc < FFN; fc += FCHK) {
        // ---- GEMM1: h[32,16] slab for this wave, K = HS.
        // Wave streams 32 consecutive 1KB tiles (32KB sequential).
        f32x8 acc1[2] = {zero, zero};
        const unsigned short* w1col =
            w1t + (((size_t)e * 256 + (fc >> 4) + wv) * 32) * TILE;
        for (int kb = 0; kb < 32; kb += 4) {
            bf16x16 bt[4];
            #pragma unroll
            for (int q = 0; q < 4; ++q)
                bt[q] = load_bt(w1col + (size_t)(kb + q) * TILE, lane);
            #pragma unroll
            for (int q = 0; q < 4; ++q) {
                int kk = (kb + q) * 32;
                bf16x16 a0 = load_a(Xlds,             XSTR, kk, lane);
                bf16x16 a1 = load_a(Xlds + 16 * XSTR, XSTR, kk, lane);
                acc1[0] = __builtin_amdgcn_wmma_f32_16x16x32_bf16(
                              false, a0, false, bt[q], (short)0, acc1[0], false, false);
                acc1[1] = __builtin_amdgcn_wmma_f32_16x16x32_bf16(
                              false, a1, false, bt[q], (short)0, acc1[1], false, false);
            }
        }

        // exact gelu, then publish slab to LDS for all waves
        #pragma unroll
        for (int h2 = 0; h2 < 2; ++h2) {
            #pragma unroll
            for (int r = 0; r < 8; ++r) {
                float v = acc1[h2][r];
                float g = 0.5f * v * (1.0f + erff(v * 0.70710678118654752f));
                int m = h2 * 16 + ((lane < 16) ? r : (r + 8));
                Hlds[m * HSTR + wv * 16 + (lane & 15)] = f2bf(g);
            }
        }
        __syncthreads();

        // ---- GEMM2: accOut[32,128] += h[32,FCHK] @ w2[fc:fc+FCHK, wv*128:...]
        // Wave streams 8 consecutive 1KB tiles (8KB sequential) per kk step.
        #pragma unroll
        for (int kk = 0; kk < FCHK; kk += 32) {
            bf16x16 a0 = load_a(Hlds,             HSTR, kk, lane);
            bf16x16 a1 = load_a(Hlds + 16 * HSTR, HSTR, kk, lane);
            const unsigned short* w2row =
                w2t + (((size_t)e * 128 + ((fc + kk) >> 5)) * 64 + wv * 8) * TILE;
            bf16x16 bt[8];
            #pragma unroll
            for (int nt = 0; nt < 8; ++nt)
                bt[nt] = load_bt(w2row + (size_t)nt * TILE, lane);
            #pragma unroll
            for (int nt = 0; nt < 8; ++nt) {
                accOut[0][nt] = __builtin_amdgcn_wmma_f32_16x16x32_bf16(
                                    false, a0, false, bt[nt], (short)0, accOut[0][nt], false, false);
                accOut[1][nt] = __builtin_amdgcn_wmma_f32_16x16x32_bf16(
                                    false, a1, false, bt[nt], (short)0, accOut[1][nt], false, false);
            }
        }
        __syncthreads();   // Hlds reused next chunk
    }

    // combine-weighted scatter (tokens can live in two experts -> f32 atomics)
    #pragma unroll
    for (int h2 = 0; h2 < 2; ++h2) {
        #pragma unroll
        for (int nt = 0; nt < 8; ++nt) {
            int col = wv * 128 + nt * 16 + (lane & 15);
            #pragma unroll
            for (int r = 0; r < 8; ++r) {
                int m = h2 * 16 + ((lane < 16) ? r : (r + 8));
                float v = accOut[h2][nt][r] * cwS[m];
                unsafeAtomicAdd(out + (size_t)tokS[m] * HSZ + col, v);
            }
        }
    }
}

// ---------------------------------------------------------------- launch
extern "C" void kernel_launch(void* const* d_in, const int* in_sizes, int n_in,
                              void* d_out, int out_size, void* d_ws, size_t ws_size,
                              hipStream_t stream) {
    const float* x    = (const float*)d_in[0];
    const float* ew   = (const float*)d_in[1];
    /* d_in[2] = scores: unused by the reference math */
    const float* w1   = (const float*)d_in[3];
    const float* w2   = (const float*)d_in[4];
    const int*   eidx = (const int*)d_in[5];
    float* out = (float*)d_out;

    char* ws = (char*)d_ws;
    size_t off = 0;
    auto carve = [&](size_t bytes) -> char* {
        char* p = ws + off;
        off = (off + bytes + 255) & ~(size_t)255;
        return p;
    };
    int*            counts = (int*)            carve(NEXP * sizeof(int));
    int*            toks   = (int*)            carve((size_t)NEXP * CAP * sizeof(int));
    float*          cwsb   = (float*)          carve((size_t)NEXP * CAP * sizeof(float));
    unsigned short* w1t    = (unsigned short*) carve((size_t)NEXP * HSZ * FFN * 2);
    unsigned short* w2t    = (unsigned short*) carve((size_t)NEXP * FFN * HSZ * 2);
    (void)ws_size; (void)in_sizes; (void)n_in; (void)out_size;

    hipMemsetAsync(counts, 0, NEXP * sizeof(int), stream);
    hipMemsetAsync(d_out, 0, (size_t)NTOK * HSZ * sizeof(float), stream);

    cvt_swz_w1_kernel<<<8192, 256, 0, stream>>>(w1, w1t);
    cvt_swz_w2_kernel<<<8192, 256, 0, stream>>>(w2, w2t);
    route_kernel<<<(NTOK * TOPK + 255) / 256, 256, 0, stream>>>(eidx, ew, counts, toks, cwsb);

    dim3 grid(NEXP, CAP / MT);
    moe_mlp_kernel<<<grid, 256, 0, stream>>>(x, counts, toks, cwsb, w1t, w2t, out);
}
